// MultiHeadAttention_89833535964017
// MI455X (gfx1250) — matrix operand order repaired and checked by reading, compiled, not run on hardware
//
#include <hip/hip_runtime.h>

// ---------------------------------------------------------------------------
// MHA forward for MI455X (gfx1250): bf16 WMMA + async-to-LDS double buffering.
//   EMBED=1024, HEADS=16, DHEAD=64, N=4, L=2048 -> M = N*L = 8192 rows.
// Pipeline: cvt(fp32->bf16) -> 3 proj GEMMs -> flash attention -> out GEMM.
// ---------------------------------------------------------------------------

#define EMBED 1024
#define HEADS 16
#define DHEAD 64
#define BATCH 4
#define SEQ   2048
#define MTOT  (BATCH * SEQ) // 8192

typedef __attribute__((ext_vector_type(16))) __bf16 v16bf;
typedef __attribute__((ext_vector_type(8)))  float  v8f;
typedef __attribute__((__vector_size__(16))) int    vi4;   // exact pointee type
                                                           // of the async-LDS builtin

union FragB16 {
    v16bf v;
    unsigned short u[16];
    uint4 q[2];
};

#ifndef __has_builtin
#define __has_builtin(x) 0
#endif

#if __has_builtin(__builtin_amdgcn_global_load_async_to_lds_b128)
#define HAVE_ASYNC_LDS 1
#else
#define HAVE_ASYNC_LDS 0
#endif

#if HAVE_ASYNC_LDS && __has_builtin(__builtin_amdgcn_s_wait_asynccnt)
#define WAIT_ASYNC() __builtin_amdgcn_s_wait_asynccnt(0)
#elif HAVE_ASYNC_LDS
#define WAIT_ASYNC() asm volatile("s_wait_asynccnt 0x0" ::: "memory")
#else
#define WAIT_ASYNC()
#endif

// 16-byte global -> LDS copy (async DMA when the toolchain exposes it)
__device__ __forceinline__ void cp16_g2l(unsigned short* l, const unsigned short* g) {
#if HAVE_ASYNC_LDS
    __builtin_amdgcn_global_load_async_to_lds_b128(
        (__attribute__((address_space(1))) vi4*)(unsigned short*)g,
        (__attribute__((address_space(3))) vi4*)l,
        0, 0);
#else
    *(uint4*)l = *(const uint4*)g;
#endif
}

// fp32 -> bf16 round-to-nearest-even
__device__ __forceinline__ unsigned short f2bf(float f) {
    unsigned int u = __float_as_uint(f);
    u += 0x7FFFu + ((u >> 16) & 1u);
    return (unsigned short)(u >> 16);
}

// ---------------------------------------------------------------------------
// Elementwise fp32 -> bf16 (bandwidth-trivial; keeps conversion VALU out of
// the GEMM hot loops and converts weights once instead of per-block).
// ---------------------------------------------------------------------------
__global__ __launch_bounds__(256)
void mha_cvt_bf16(const float* __restrict__ src, unsigned short* __restrict__ dst, int n4)
{
    const int i = blockIdx.x * 256 + threadIdx.x;
    if (i < n4) {
        const float4 f = ((const float4*)src)[i];
        uint2 p;
        p.x = (unsigned)f2bf(f.x) | ((unsigned)f2bf(f.y) << 16);
        p.y = (unsigned)f2bf(f.z) | ((unsigned)f2bf(f.w) << 16);
        ((uint2*)dst)[i] = p;
    }
}

// ---------------------------------------------------------------------------
// GEMM:  Out[m,n] = sum_k A[m,k] * W[n,k] + bias[n]      (A, W already bf16)
//   mode 0: bf16 row-major [m, n]
//   mode 1: bf16 transposed per head:  [((b*16+h)*64 + d) * 2048 + l]
//   mode 2: fp32 row-major [m, n]
// Block tile 128x64, 8 waves each 32x32. K-stage 64 (8 WMMA / wave / stage),
// double-buffered LDS fed by async global->LDS copies; one barrier per stage.
// ---------------------------------------------------------------------------
__global__ __launch_bounds__(256)
void mha_gemm_wmma(const unsigned short* __restrict__ A,
                   const unsigned short* __restrict__ Wb,
                   const float* __restrict__ bias,
                   unsigned short* __restrict__ OutBF,
                   float* __restrict__ OutF,
                   int mode)
{
    __shared__ __align__(16) unsigned short As[2][128][64]; // 32 KB
    __shared__ __align__(16) unsigned short Bs[2][64][64];  // 16 KB

    const int tid  = threadIdx.x;
    const int lane = tid & 31;
    const int lo   = lane & 15;
    const int hi   = lane >> 4;
    const int wid  = tid >> 5;
    const int r0   = (wid & 3) * 32;
    const int n0   = (wid >> 2) * 32;
    const int gRow0 = blockIdx.x * 128;
    const int gCol0 = blockIdx.y * 64;

    v8f acc00 = 0.0f, acc01 = 0.0f, acc10 = 0.0f, acc11 = 0.0f;

    auto stage = [&](int buf, int kc) {
        #pragma unroll
        for (int i = 0; i < 4; ++i) {           // A: 128x64 bf16 = 1024 16B chunks
            const int idx = tid + i * 256;      // 0..1023
            const int row = idx >> 3;           // 0..127
            const int c8  = (idx & 7) * 8;      // 0..56
            cp16_g2l(&As[buf][row][c8],
                     &A[(size_t)(gRow0 + row) * EMBED + kc + c8]);
        }
        #pragma unroll
        for (int i = 0; i < 2; ++i) {           // B: 64x64 bf16 = 512 16B chunks
            const int idx = tid + i * 256;      // 0..511
            const int row = idx >> 3;           // 0..63
            const int c8  = (idx & 7) * 8;
            cp16_g2l(&Bs[buf][row][c8],
                     &Wb[(size_t)(gCol0 + row) * EMBED + kc + c8]);
        }
    };

    stage(0, 0);

    for (int s = 0; s < EMBED / 64; ++s) {
        WAIT_ASYNC();           // own async copies for stage s have landed
        __syncthreads();        // every wave's copies landed; prev reads done
        if (s + 1 < EMBED / 64) stage((s + 1) & 1, (s + 1) * 64);

        const int buf = s & 1;
        #pragma unroll
        for (int kk = 0; kk < 64; kk += 32) {
            FragB16 a0, a1, b0, b1;
            a0.q[0] = *(const uint4*)&As[buf][r0 + lo][kk + hi * 8];
            a0.q[1] = *(const uint4*)&As[buf][r0 + lo][kk + 16 + hi * 8];
            a1.q[0] = *(const uint4*)&As[buf][r0 + 16 + lo][kk + hi * 8];
            a1.q[1] = *(const uint4*)&As[buf][r0 + 16 + lo][kk + 16 + hi * 8];
            b0.q[0] = *(const uint4*)&Bs[buf][n0 + lo][kk + hi * 8];
            b0.q[1] = *(const uint4*)&Bs[buf][n0 + lo][kk + 16 + hi * 8];
            b1.q[0] = *(const uint4*)&Bs[buf][n0 + 16 + lo][kk + hi * 8];
            b1.q[1] = *(const uint4*)&Bs[buf][n0 + 16 + lo][kk + 16 + hi * 8];

            acc00 = __builtin_amdgcn_wmma_f32_16x16x32_bf16(false, a0.v, false, b0.v, (short)0, acc00, false, false);
            acc01 = __builtin_amdgcn_wmma_f32_16x16x32_bf16(false, a0.v, false, b1.v, (short)0, acc01, false, false);
            acc10 = __builtin_amdgcn_wmma_f32_16x16x32_bf16(false, a1.v, false, b0.v, (short)0, acc10, false, false);
            acc11 = __builtin_amdgcn_wmma_f32_16x16x32_bf16(false, a1.v, false, b1.v, (short)0, acc11, false, false);
        }
    }

    // ---- epilogue: bias add + store ----
    #pragma unroll
    for (int i = 0; i < 2; ++i) {
        #pragma unroll
        for (int j = 0; j < 2; ++j) {
            v8f acv = (i == 0) ? (j == 0 ? acc00 : acc01)
                               : (j == 0 ? acc10 : acc11);
            const int nn = gCol0 + n0 + j * 16 + lo;
            const float bval = bias[nn];
            const int mb = gRow0 + r0 + i * 16 + hi * 8;
            #pragma unroll
            for (int r = 0; r < 8; ++r) {
                const int m = mb + r;
                const float val = acv[r] + bval;
                if (mode == 0) {
                    OutBF[(size_t)m * EMBED + nn] = f2bf(val);
                } else if (mode == 1) {
                    const int b = m >> 11, l = m & 2047;
                    const int h = nn >> 6, d = nn & 63;
                    OutBF[((size_t)((b * HEADS + h) * DHEAD + d)) * SEQ + l] = f2bf(val);
                } else {
                    OutF[(size_t)m * EMBED + nn] = val;
                }
            }
        }
    }
}

// ---------------------------------------------------------------------------
// Flash attention (masks are dead in the reference; softmax scale 1/32).
//   QP, KP: bf16 [8192, 1024] row-major; VT: bf16 [(b*16+h)*64+d][2048].
// Wave owns 16 q rows; key loop in blocks of 32. P staging LDS is per-wave,
// and same-wave LDS requests are processed in order (ISA 7.3), so only
// compiler memory barriers are needed -- no workgroup barriers at all.
// ---------------------------------------------------------------------------
__global__ __launch_bounds__(256)
void mha_flash_wmma(const unsigned short* __restrict__ QP,
                    const unsigned short* __restrict__ KP,
                    const unsigned short* __restrict__ VT,
                    unsigned short* __restrict__ CTX)
{
    __shared__ __align__(16) unsigned short Ps[8][16][32]; // per-wave P tile

    const int tid  = threadIdx.x;
    const int lane = tid & 31;
    const int lo   = lane & 15;
    const int hi   = lane >> 4;
    const int wid  = tid >> 5;

    const int bh = blockIdx.y;
    const int b  = bh >> 4;
    const int h  = bh & 15;
    const int q0 = blockIdx.x * 128 + wid * 16;

    FragB16 aq0, aq1; // Q A-fragments for d-chunks [0,32), [32,64)
    {
        const size_t base = ((size_t)(b * SEQ + q0 + lo)) * EMBED + h * DHEAD;
        aq0.q[0] = *(const uint4*)&QP[base + hi * 8];
        aq0.q[1] = *(const uint4*)&QP[base + 16 + hi * 8];
        aq1.q[0] = *(const uint4*)&QP[base + 32 + hi * 8];
        aq1.q[1] = *(const uint4*)&QP[base + 32 + 16 + hi * 8];
    }

    float m_run[8], l_run[8];
    #pragma unroll
    for (int r = 0; r < 8; ++r) { m_run[r] = -1e30f; l_run[r] = 0.0f; }
    v8f o0 = 0.0f, o1 = 0.0f, o2 = 0.0f, o3 = 0.0f;

    const float sc = 0.03125f * 1.44269504088896f; // (1/sqrt(1024)) * log2(e)

    for (int kb = 0; kb < SEQ; kb += 32) {
        // ---- S = Q @ K^T for 32 keys: two 16x16 f32 tiles ----
        v8f s0 = 0.0f, s1 = 0.0f;
        {
            const size_t k0 = ((size_t)(b * SEQ + kb + lo)) * EMBED + h * DHEAD;
            const size_t k1 = ((size_t)(b * SEQ + kb + 16 + lo)) * EMBED + h * DHEAD;
            FragB16 bk;
            bk.q[0] = *(const uint4*)&KP[k0 + hi * 8];
            bk.q[1] = *(const uint4*)&KP[k0 + 16 + hi * 8];
            s0 = __builtin_amdgcn_wmma_f32_16x16x32_bf16(false, aq0.v, false, bk.v, (short)0, s0, false, false);
            bk.q[0] = *(const uint4*)&KP[k0 + 32 + hi * 8];
            bk.q[1] = *(const uint4*)&KP[k0 + 32 + 16 + hi * 8];
            s0 = __builtin_amdgcn_wmma_f32_16x16x32_bf16(false, aq1.v, false, bk.v, (short)0, s0, false, false);
            bk.q[0] = *(const uint4*)&KP[k1 + hi * 8];
            bk.q[1] = *(const uint4*)&KP[k1 + 16 + hi * 8];
            s1 = __builtin_amdgcn_wmma_f32_16x16x32_bf16(false, aq0.v, false, bk.v, (short)0, s1, false, false);
            bk.q[0] = *(const uint4*)&KP[k1 + 32 + hi * 8];
            bk.q[1] = *(const uint4*)&KP[k1 + 32 + 16 + hi * 8];
            s1 = __builtin_amdgcn_wmma_f32_16x16x32_bf16(false, aq1.v, false, bk.v, (short)0, s1, false, false);
        }

        // ---- online softmax update (row r lives in a 16-lane half-group) ----
        #pragma unroll
        for (int r = 0; r < 8; ++r) {
            float smax = fmaxf(s0[r], s1[r]);
            smax = fmaxf(smax, __shfl_xor(smax, 1, 32));
            smax = fmaxf(smax, __shfl_xor(smax, 2, 32));
            smax = fmaxf(smax, __shfl_xor(smax, 4, 32));
            smax = fmaxf(smax, __shfl_xor(smax, 8, 32));
            const float mnew  = fmaxf(m_run[r], smax);
            const float alpha = exp2f((m_run[r] - mnew) * sc);
            const float p0 = exp2f((s0[r] - mnew) * sc);
            const float p1 = exp2f((s1[r] - mnew) * sc);
            float rs = p0 + p1;
            rs += __shfl_xor(rs, 1, 32);
            rs += __shfl_xor(rs, 2, 32);
            rs += __shfl_xor(rs, 4, 32);
            rs += __shfl_xor(rs, 8, 32);
            l_run[r] = l_run[r] * alpha + rs;
            m_run[r] = mnew;
            o0[r] *= alpha; o1[r] *= alpha; o2[r] *= alpha; o3[r] *= alpha;
            Ps[wid][r + hi * 8][lo]      = f2bf(p0);
            Ps[wid][r + hi * 8][16 + lo] = f2bf(p1);
        }
        asm volatile("" ::: "memory"); // keep ds_store before ds_load; HW is in-order

        FragB16 pa; // P as A-fragment (rows = q, K-dim = 32 keys)
        pa.q[0] = *(const uint4*)&Ps[wid][lo][hi * 8];
        pa.q[1] = *(const uint4*)&Ps[wid][lo][16 + hi * 8];
        asm volatile("" ::: "memory"); // keep next iter's stores after these loads

        // ---- O += P @ V  (4 d-chunks of 16) ----
        {
            const size_t vb = ((size_t)((b * HEADS + h) * DHEAD + lo)) * SEQ + kb + hi * 8;
            FragB16 bv;
            bv.q[0] = *(const uint4*)&VT[vb + 0 * 16 * SEQ];
            bv.q[1] = *(const uint4*)&VT[vb + 0 * 16 * SEQ + 16];
            o0 = __builtin_amdgcn_wmma_f32_16x16x32_bf16(false, pa.v, false, bv.v, (short)0, o0, false, false);
            bv.q[0] = *(const uint4*)&VT[vb + 1 * 16 * SEQ];
            bv.q[1] = *(const uint4*)&VT[vb + 1 * 16 * SEQ + 16];
            o1 = __builtin_amdgcn_wmma_f32_16x16x32_bf16(false, pa.v, false, bv.v, (short)0, o1, false, false);
            bv.q[0] = *(const uint4*)&VT[vb + 2 * 16 * SEQ];
            bv.q[1] = *(const uint4*)&VT[vb + 2 * 16 * SEQ + 16];
            o2 = __builtin_amdgcn_wmma_f32_16x16x32_bf16(false, pa.v, false, bv.v, (short)0, o2, false, false);
            bv.q[0] = *(const uint4*)&VT[vb + 3 * 16 * SEQ];
            bv.q[1] = *(const uint4*)&VT[vb + 3 * 16 * SEQ + 16];
            o3 = __builtin_amdgcn_wmma_f32_16x16x32_bf16(false, pa.v, false, bv.v, (short)0, o3, false, false);
        }
    }

    // ---- normalize and store context (bf16 row-major) ----
    #pragma unroll
    for (int r = 0; r < 8; ++r) {
        const float inv = 1.0f / l_run[r];
        const size_t row = (size_t)(b * SEQ + q0 + r + hi * 8) * EMBED + h * DHEAD + lo;
        CTX[row + 0]  = f2bf(o0[r] * inv);
        CTX[row + 16] = f2bf(o1[r] * inv);
        CTX[row + 32] = f2bf(o2[r] * inv);
        CTX[row + 48] = f2bf(o3[r] * inv);
    }
}

// ---------------------------------------------------------------------------
// Workspace: qb|kb|vb|QP|KP|VT|CTX (16 MB each) + Wqb|Wkb|Wvb|Wob (2 MB each)
// = 120 MB total.
// ---------------------------------------------------------------------------
extern "C" void kernel_launch(void* const* d_in, const int* in_sizes, int n_in,
                              void* d_out, int out_size, void* d_ws, size_t ws_size,
                              hipStream_t stream) {
    const float* q  = (const float*)d_in[0];
    const float* k  = (const float*)d_in[1];
    const float* v  = (const float*)d_in[2];
    // d_in[3] padding_mask, d_in[4] sequence_mask: dead in reference.
    const float* Wq = (const float*)d_in[5];
    const float* bq = (const float*)d_in[6];
    const float* Wk = (const float*)d_in[7];
    const float* bk = (const float*)d_in[8];
    const float* Wv = (const float*)d_in[9];
    const float* bv = (const float*)d_in[10];
    const float* Wo = (const float*)d_in[11];
    const float* bo = (const float*)d_in[12];
    float* out = (float*)d_out;

    char* ws = (char*)d_ws;
    const size_t BIG = (size_t)MTOT * EMBED * sizeof(unsigned short);   // 16 MB
    const size_t SML = (size_t)EMBED * EMBED * sizeof(unsigned short);  //  2 MB
    unsigned short* qb  = (unsigned short*)(ws + 0 * BIG);
    unsigned short* kb  = (unsigned short*)(ws + 1 * BIG);
    unsigned short* vb  = (unsigned short*)(ws + 2 * BIG);
    unsigned short* QP  = (unsigned short*)(ws + 3 * BIG);
    unsigned short* KP  = (unsigned short*)(ws + 4 * BIG);
    unsigned short* VT  = (unsigned short*)(ws + 5 * BIG);
    unsigned short* CTX = (unsigned short*)(ws + 6 * BIG);
    unsigned short* Wqb = (unsigned short*)(ws + 7 * BIG + 0 * SML);
    unsigned short* Wkb = (unsigned short*)(ws + 7 * BIG + 1 * SML);
    unsigned short* Wvb = (unsigned short*)(ws + 7 * BIG + 2 * SML);
    unsigned short* Wob = (unsigned short*)(ws + 7 * BIG + 3 * SML);

    const int n4_big = MTOT * EMBED / 4;   // 2M float4 chunks
    const int n4_sml = EMBED * EMBED / 4;  // 256K
    mha_cvt_bf16<<<n4_big / 256, 256, 0, stream>>>(q, qb, n4_big);
    mha_cvt_bf16<<<n4_big / 256, 256, 0, stream>>>(k, kb, n4_big);
    mha_cvt_bf16<<<n4_big / 256, 256, 0, stream>>>(v, vb, n4_big);
    mha_cvt_bf16<<<n4_sml / 256, 256, 0, stream>>>(Wq, Wqb, n4_sml);
    mha_cvt_bf16<<<n4_sml / 256, 256, 0, stream>>>(Wk, Wkb, n4_sml);
    mha_cvt_bf16<<<n4_sml / 256, 256, 0, stream>>>(Wv, Wvb, n4_sml);
    mha_cvt_bf16<<<n4_sml / 256, 256, 0, stream>>>(Wo, Wob, n4_sml);

    const dim3 gg(MTOT / 128, EMBED / 64);
    mha_gemm_wmma<<<gg, 256, 0, stream>>>(qb, Wqb, bq, QP, nullptr, 0);
    mha_gemm_wmma<<<gg, 256, 0, stream>>>(kb, Wkb, bk, KP, nullptr, 0);
    mha_gemm_wmma<<<gg, 256, 0, stream>>>(vb, Wvb, bv, VT, nullptr, 1);
    mha_flash_wmma<<<dim3(SEQ / 128, BATCH * HEADS), 256, 0, stream>>>(QP, KP, VT, CTX);
    mha_gemm_wmma<<<gg, 256, 0, stream>>>(CTX, Wob, bo, nullptr, out, 2);
}
